// InfGCN_55009941127335
// MI455X (gfx1250) — compile-verified, matrix operand order checked by reading, not run on hardware
//
#include <hip/hip_runtime.h>
#include <cmath>

typedef __attribute__((ext_vector_type(16))) _Float16 v16h;
typedef __attribute__((ext_vector_type(8)))  float    v8f;

#define WAVES_PER_BLOCK 8
#define RS3 0.5773502691896258f   /* 1/sqrt(3) */
#define RS2 0.7071067811865476f   /* 1/sqrt(2) */
#define INV_MUL 0.25f             /* 1/sqrt(16) */

static __device__ __forceinline__ v8f wmma_f16(v16h a, v16h b, v8f c) {
    // D = A(16x32 f16) * B(32x16 f16) + C(16x16 f32)
    return __builtin_amdgcn_wmma_f32_16x16x32_f16(false, a, false, b,
                                                  (short)0, c, false, false);
}

static __device__ __forceinline__ float silu_f(float x) {
    return x / (1.0f + __expf(-x));
}

// ---------------------------------------------------------------------------
// Pack a K x N f32 weight matrix into WMMA-B fragment order (f16), scale
// folded in.  Tile t = n0*(K/32) + kt holds the 32x16 block (kt,n0); within a
// tile, element index = lane*16 + h, with the ISA 16-bit B layout:
//   N = n0*16 + (lane&15)
//   K = kt*32 + (h&7) + ((h>>3)<<4) + (((lane>>4)&1)<<3)
// ---------------------------------------------------------------------------
__global__ __launch_bounds__(256) void pack_w_kernel(const float* __restrict__ W,
                                                     _Float16* __restrict__ outp,
                                                     int K, int N, int total,
                                                     float scale) {
    int tid = blockIdx.x * 256 + threadIdx.x;
    if (tid >= total) return;
    int tile = tid >> 9;
    int rem  = tid & 511;
    int lane = rem >> 4;
    int h    = rem & 15;
    int ktiles = K >> 5;
    int n0 = tile / ktiles;
    int kt = tile - n0 * ktiles;
    int n  = n0 * 16 + (lane & 15);
    int kk = (h & 7) + ((h >> 3) << 4) + (((lane >> 4) & 1) << 3);
    int k  = kt * 32 + kk;
    outp[tid] = (_Float16)(W[(size_t)k * N + n] * scale);
}

// ---------------------------------------------------------------------------
// Self-connection: out[n, :] = concat(x0 @ sc_w0, einsum('ud,uw->wd')) / 4
// Also serves as the d_out initializer (plain stores).
// ---------------------------------------------------------------------------
__global__ __launch_bounds__(256) void selfconn_kernel(const float* __restrict__ nf,
                                                       const float* __restrict__ w0,
                                                       const float* __restrict__ w1,
                                                       float* __restrict__ out,
                                                       int N) {
    int tid = blockIdx.x * 256 + threadIdx.x;
    if (tid >= N * 64) return;
    int n = tid >> 6;
    int j = tid & 63;
    const float* r = nf + (size_t)n * 64;
    float s = 0.0f;
    if (j < 16) {
        #pragma unroll
        for (int u = 0; u < 16; ++u) s += r[u] * w0[u * 16 + j];
    } else {
        int jj = j - 16;
        int wc = jj / 3;
        int d  = jj - wc * 3;
        #pragma unroll
        for (int u = 0; u < 16; ++u) s += r[16 + u * 3 + d] * w1[u * 16 + wc];
    }
    out[tid] = s * INV_MUL;
}

// ---------------------------------------------------------------------------
// Fused edge kernel: radial MLP (WMMA f16) + tensor product + scatter-add.
// One wave owns 16 edges; 8 waves per block.
// ---------------------------------------------------------------------------
__global__ __launch_bounds__(256) void edge_kernel(const int* __restrict__ eidx,
                                                   const float* __restrict__ node_feat,
                                                   const float* __restrict__ edge_feat,
                                                   const float* __restrict__ edge_embed,
                                                   const _Float16* __restrict__ w1p,
                                                   const _Float16* __restrict__ w2p,
                                                   const _Float16* __restrict__ w3p,
                                                   float* __restrict__ out,
                                                   int E) {
    // per-wave private LDS regions (no cross-wave sharing -> no block barriers)
    __shared__ __align__(32) _Float16 sh1[WAVES_PER_BLOCK][2048]; // h1 in A-frag order
    __shared__ __align__(32) _Float16 sh2[WAVES_PER_BLOCK][2048]; // h2 in A-frag order
    __shared__ float scoef[WAVES_PER_BLOCK][9][16][16];           // [coef][edge][u]
    __shared__ float sy1[WAVES_PER_BLOCK][16][3];
    __shared__ int   sdst[WAVES_PER_BLOCK][16];

    const int lane   = threadIdx.x & 31;
    const int wave   = threadIdx.x >> 5;
    const int eBase  = (blockIdx.x * WAVES_PER_BLOCK + wave) * 16;
    if (eBase + 16 > E) return;                  // wave-uniform exit

    const int wcol   = lane & 15;                // N / M-across-lanes index
    const int rowoff = (lane >> 4) << 3;         // +8 for upper half-wave

    // ---- per-edge tensor-product coefficients -> LDS ----------------------
    {
        int m = wcol;
        int e = eBase + m;
        int srcn = eidx[e];
        int dstn = eidx[E + e];
        float y0  = edge_feat[(size_t)e * 4 + 0];
        float y1x = edge_feat[(size_t)e * 4 + 1];
        float y1y = edge_feat[(size_t)e * 4 + 2];
        float y1z = edge_feat[(size_t)e * 4 + 3];
        if (lane < 16) {
            sdst[wave][m] = dstn;
            sy1[wave][m][0] = y1x; sy1[wave][m][1] = y1y; sy1[wave][m][2] = y1z;
        }
        const float* nf = node_feat + (size_t)srcn * 64;
        int u0 = rowoff;                         // lanes 0-15: u 0..7, 16-31: u 8..15
        #pragma unroll
        for (int uu = 0; uu < 8; ++uu) {
            int u = u0 + uu;
            float x0 = nf[u];
            float xx = nf[16 + u * 3 + 0];
            float xy = nf[16 + u * 3 + 1];
            float xz = nf[16 + u * 3 + 2];
            scoef[wave][0][m][u] = x0 * y0;                                   // path 0
            scoef[wave][1][m][u] = (xx * y1x + xy * y1y + xz * y1z) * RS3;    // path 1
            scoef[wave][2][m][u] = x0;                                        // path 2
            scoef[wave][3][m][u] = xx * y0;                                   // path 3
            scoef[wave][4][m][u] = xy * y0;
            scoef[wave][5][m][u] = xz * y0;
            scoef[wave][6][m][u] = (xy * y1z - xz * y1y) * RS2;               // path 4
            scoef[wave][7][m][u] = (xz * y1x - xx * y1z) * RS2;
            scoef[wave][8][m][u] = (xx * y1y - xy * y1x) * RS2;
        }
    }

    // ---- build A fragments for layer 1 from edge_embed (f32 -> f16) -------
    v16h a0, a1;
    {
        const float* rowp = edge_embed + (size_t)(eBase + wcol) * 64;
        #pragma unroll
        for (int h = 0; h < 16; ++h) {
            int kk = (h & 7) + ((h >> 3) << 4) + rowoff;  // ISA 16-bit A K-pattern
            a0[h] = (_Float16)rowp[kk];
            a1[h] = (_Float16)rowp[32 + kk];
        }
    }

    // ---- layer 1: (16x64) @ (64x128), silu, store A-frag-ordered to LDS ---
    #pragma unroll
    for (int n0 = 0; n0 < 8; ++n0) {
        v8f acc = {};
        v16h b0 = *(const v16h*)(w1p + (size_t)(n0 * 2 + 0) * 512 + lane * 16);
        acc = wmma_f16(a0, b0, acc);
        v16h b1 = *(const v16h*)(w1p + (size_t)(n0 * 2 + 1) * 512 + lane * 16);
        acc = wmma_f16(a1, b1, acc);

        int Ncol = n0 * 16 + wcol;               // k-index of next layer
        int kt   = Ncol >> 5;
        int kk2  = Ncol & 31;
        int hh   = (kk2 & 7) + ((kk2 & 16) ? 8 : 0);
        int hi2  = (kk2 >> 3) & 1;
        _Float16* dst0 = &sh1[wave][kt * 512 + hi2 * 256 + hh];
        #pragma unroll
        for (int r = 0; r < 8; ++r) {
            int M = r + rowoff;
            dst0[M * 16] = (_Float16)silu_f(acc[r]);
        }
    }
    asm volatile("s_wait_dscnt 0" ::: "memory");

    // ---- layer 2: (16x128) @ (128x128), silu ------------------------------
    v16h A1f[4];
    #pragma unroll
    for (int kt = 0; kt < 4; ++kt)
        A1f[kt] = *(const v16h*)&sh1[wave][kt * 512 + lane * 16];

    #pragma unroll
    for (int n0 = 0; n0 < 8; ++n0) {
        v8f acc = {};
        #pragma unroll
        for (int kt = 0; kt < 4; ++kt) {
            v16h b = *(const v16h*)(w2p + (size_t)(n0 * 4 + kt) * 512 + lane * 16);
            acc = wmma_f16(A1f[kt], b, acc);
        }
        int Ncol = n0 * 16 + wcol;
        int kt   = Ncol >> 5;
        int kk2  = Ncol & 31;
        int hh   = (kk2 & 7) + ((kk2 & 16) ? 8 : 0);
        int hi2  = (kk2 >> 3) & 1;
        _Float16* dst0 = &sh2[wave][kt * 512 + hi2 * 256 + hh];
        #pragma unroll
        for (int r = 0; r < 8; ++r) {
            int M = r + rowoff;
            dst0[M * 16] = (_Float16)silu_f(acc[r]);
        }
    }
    asm volatile("s_wait_dscnt 0" ::: "memory");

    // ---- layer 3 + tensor product: stream 80 n-tiles of w -----------------
    v16h A2f[4];
    #pragma unroll
    for (int kt = 0; kt < 4; ++kt)
        A2f[kt] = *(const v16h*)&sh2[wave][kt * 512 + lane * 16];

    const float* cf = &scoef[wave][0][0][0];
    v8f o0 = {}, t2 = {}, o1a = {}, o1b = {}, o1c = {};

#define COMPUTE_D(n0)                                                          \
    {                                                                          \
        d = (v8f){};                                                           \
        __builtin_prefetch(w3p + (size_t)((n0) + 1) * 2048 + lane * 16, 0, 0); \
        _Pragma("unroll")                                                      \
        for (int kt = 0; kt < 4; ++kt) {                                       \
            v16h b = *(const v16h*)(w3p + (size_t)((n0) * 4 + kt) * 512 +      \
                                    lane * 16);                                \
            d = wmma_f16(A2f[kt], b, d);                                       \
        }                                                                      \
    }

    v8f d;
    // paths 0,1 -> o0
    for (int pu = 0; pu < 32; ++pu) {
        COMPUTE_D(pu);
        int c = pu >> 4, u = pu & 15;
        #pragma unroll
        for (int r = 0; r < 8; ++r)
            o0[r] += d[r] * cf[(c * 16 + (r + rowoff)) * 16 + u];
    }
    // path 2 -> t2 (later outer-product with y1)
    for (int u = 0; u < 16; ++u) {
        COMPUTE_D(32 + u);
        #pragma unroll
        for (int r = 0; r < 8; ++r)
            t2[r] += d[r] * cf[(2 * 16 + (r + rowoff)) * 16 + u];
    }
    // path 3 -> o1 (x1 * y0 coefficients, per component)
    for (int u = 0; u < 16; ++u) {
        COMPUTE_D(48 + u);
        #pragma unroll
        for (int r = 0; r < 8; ++r) {
            float dv = d[r];
            int rm = r + rowoff;
            o1a[r] += dv * cf[(3 * 16 + rm) * 16 + u];
            o1b[r] += dv * cf[(4 * 16 + rm) * 16 + u];
            o1c[r] += dv * cf[(5 * 16 + rm) * 16 + u];
        }
    }
    // path 4 -> o1 (cross / sqrt2 coefficients)
    for (int u = 0; u < 16; ++u) {
        COMPUTE_D(64 + u);
        #pragma unroll
        for (int r = 0; r < 8; ++r) {
            float dv = d[r];
            int rm = r + rowoff;
            o1a[r] += dv * cf[(6 * 16 + rm) * 16 + u];
            o1b[r] += dv * cf[(7 * 16 + rm) * 16 + u];
            o1c[r] += dv * cf[(8 * 16 + rm) * 16 + u];
        }
    }
#undef COMPUTE_D

    // ---- finalize + scatter-add ------------------------------------------
    #pragma unroll
    for (int r = 0; r < 8; ++r) {
        int m  = r + rowoff;
        int dn = sdst[wave][m];
        float* ob = out + (size_t)dn * 64;
        float yl0 = sy1[wave][m][0], yl1 = sy1[wave][m][1], yl2 = sy1[wave][m][2];
        unsafeAtomicAdd(ob + wcol, o0[r] * INV_MUL);
        unsafeAtomicAdd(ob + 16 + wcol * 3 + 0, (o1a[r] + t2[r] * yl0) * INV_MUL);
        unsafeAtomicAdd(ob + 16 + wcol * 3 + 1, (o1b[r] + t2[r] * yl1) * INV_MUL);
        unsafeAtomicAdd(ob + 16 + wcol * 3 + 2, (o1c[r] + t2[r] * yl2) * INV_MUL);
    }
}

// ---------------------------------------------------------------------------
extern "C" void kernel_launch(void* const* d_in, const int* in_sizes, int n_in,
                              void* d_out, int out_size, void* d_ws, size_t ws_size,
                              hipStream_t stream) {
    const int*   eidx       = (const int*)d_in[0];
    const float* node_feat  = (const float*)d_in[1];
    const float* edge_feat  = (const float*)d_in[2];
    const float* edge_embed = (const float*)d_in[3];
    const float* fc_w1      = (const float*)d_in[4];
    const float* fc_w2      = (const float*)d_in[5];
    const float* fc_w3      = (const float*)d_in[6];
    const float* sc_w0      = (const float*)d_in[7];
    const float* sc_w1      = (const float*)d_in[8];
    float* out = (float*)d_out;

    const int E = in_sizes[0] / 2;
    const int N = in_sizes[1] / 64;

    // packed f16 weights in workspace: w1 16 tiles, w2 32 tiles, w3 320 tiles
    _Float16* w1p = (_Float16*)d_ws;
    _Float16* w2p = w1p + 16 * 512;
    _Float16* w3p = w2p + 32 * 512;

    const float s1 = 0.125f;                  // 1/sqrt(64)
    const float s2 = 0.08838834764831845f;    // 1/sqrt(128)

    pack_w_kernel<<<(16 * 512 + 255) / 256, 256, 0, stream>>>(fc_w1, w1p, 64, 128, 16 * 512, s1);
    pack_w_kernel<<<(32 * 512 + 255) / 256, 256, 0, stream>>>(fc_w2, w2p, 128, 128, 32 * 512, s2);
    pack_w_kernel<<<(320 * 512 + 255) / 256, 256, 0, stream>>>(fc_w3, w3p, 128, 1280, 320 * 512, s2);

    selfconn_kernel<<<(N * 64 + 255) / 256, 256, 0, stream>>>(node_feat, sc_w0, sc_w1, out, N);

    int blocks = (E + 16 * WAVES_PER_BLOCK - 1) / (16 * WAVES_PER_BLOCK);
    edge_kernel<<<blocks, 256, 0, stream>>>(eidx, node_feat, edge_feat, edge_embed,
                                            w1p, w2p, w3p, out, E);
}
